// IntelligentHFTModel_73478300500054
// MI455X (gfx1250) — compile-verified, matrix operand order checked by reading, not run on hardware
//
#include <hip/hip_runtime.h>
#include <hip/hip_bf16.h>

typedef _Float16 half8 __attribute__((ext_vector_type(8)));
typedef _Float16 v16h  __attribute__((ext_vector_type(16)));
typedef float    v8f   __attribute__((ext_vector_type(8)));

#define Bc   16
#define Sc   1024
#define Fc   128
#define Dc   512
#define Hc   8
#define Lc   6
#define HDc  64
#define DFFc 1024
#define Mtok (Bc * Sc)   // 16384 tokens

// ---------------------------------------------------------------------------
// WMMA helpers.
// 16-bit A-operand fragment layout (CDNA5 ISA 7.12.2): lane r in [0,15] holds
// row r with K = {0..7, 16..23}; lanes 16..31 hold the same rows with
// K = {8..15, 24..31}.  So a fragment is two contiguous 16-byte loads per lane
// at byte offsets (g*8)*2 and (16+g*8)*2 from the row start.  We store the
// B matrix transposed (N-major) so B-fragments use the identical pattern.
// ---------------------------------------------------------------------------
__device__ __forceinline__ v16h load_frag(const _Float16* base, int ld,
                                          int row16, int k0, int lane) {
  const int r = lane & 15, g = lane >> 4;
  const _Float16* p = base + (size_t)(row16 + r) * ld + k0 + g * 8;
  half8 lo = *(const half8*)(p);
  half8 hi = *(const half8*)(p + 16);
  return __builtin_shufflevector(lo, hi, 0,1,2,3,4,5,6,7,8,9,10,11,12,13,14,15);
}

__device__ __forceinline__ v8f wmma_f32_f16(v16h a, v16h b, v8f c) {
  return __builtin_amdgcn_wmma_f32_16x16x32_f16(false, a, false, b,
                                                (short)0, c, false, false);
}

// ---------------------------------------------------------------------------
// GEMM: C(MxN) = act(A(MxK,f16) * Bt(NxK,f16)^T + bias).
// Block = 256 threads = 8 waves (2 along M x 4 along N), wave tile 64x64,
// block tile 128x256.  16 v_wmma per 32-wide K step per wave.
// ---------------------------------------------------------------------------
template <int RELU, int DO16, int DO32>
__global__ __launch_bounds__(256) void gemm_f16_wmma(
    const _Float16* __restrict__ A, const _Float16* __restrict__ Bt,
    const float* __restrict__ bias, float* __restrict__ C32,
    _Float16* __restrict__ C16, int M, int N, int K) {
  const int lane = threadIdx.x & 31;
  const int wave = threadIdx.x >> 5;
  const int wm = wave & 1;
  const int wn = wave >> 1;
  const int m0 = blockIdx.x * 128 + wm * 64;
  const int n0 = blockIdx.y * 256 + wn * 64;

  v8f acc[4][4] = {};
  for (int k0 = 0; k0 < K; k0 += 32) {
    v16h af[4], bf[4];
#pragma unroll
    for (int i = 0; i < 4; i++) af[i] = load_frag(A,  K, m0 + i * 16, k0, lane);
#pragma unroll
    for (int j = 0; j < 4; j++) bf[j] = load_frag(Bt, K, n0 + j * 16, k0, lane);
#pragma unroll
    for (int i = 0; i < 4; i++)
#pragma unroll
      for (int j = 0; j < 4; j++)
        acc[i][j] = wmma_f32_f16(af[i], bf[j], acc[i][j]);
  }

  const int col = lane & 15, g = lane >> 4;
#pragma unroll
  for (int j = 0; j < 4; j++) {
    const int n = n0 + j * 16 + col;
    const float bv = bias[n];
#pragma unroll
    for (int i = 0; i < 4; i++) {
#pragma unroll
      for (int e = 0; e < 8; e++) {
        const int m = m0 + i * 16 + e + 8 * g;       // C/D layout: M = e + 8*g
        float v = acc[i][j][e] + bv;
        if (RELU) v = fmaxf(v, 0.0f);
        const size_t idx = (size_t)m * N + n;
        if (DO32) C32[idx] = v;
        if (DO16) C16[idx] = (_Float16)v;
      }
    }
  }
}

// ---------------------------------------------------------------------------
// Attention: one wave per (batch, head, 16-query tile).
// scores (16x1024) kept as f32 in 64KB LDS; exact softmax; f32 map written to
// d_out; strip converted to f16 in-place; attn@V by WMMA with pre-transposed V.
// ---------------------------------------------------------------------------
__global__ __launch_bounds__(32) void attn_wmma(
    const _Float16* __restrict__ Q, const _Float16* __restrict__ Km,
    const _Float16* __restrict__ Vt, float* __restrict__ attnL,
    _Float16* __restrict__ AO) {
  __shared__ float sc[16 * Sc];                        // 64 KB
  const int lane = threadIdx.x;
  const int qt = blockIdx.x, head = blockIdx.y, b = blockIdx.z;
  const int col = lane & 15, g = lane >> 4;

  const _Float16* qbase = Q  + (size_t)b * Sc * Dc + head * HDc;
  const _Float16* kbase = Km + (size_t)b * Sc * Dc + head * HDc;

  const v16h qf0 = load_frag(qbase, Dc, qt * 16, 0,  lane);
  const v16h qf1 = load_frag(qbase, Dc, qt * 16, 32, lane);

  for (int kt = 0; kt < Sc / 16; kt++) {
    v16h kf0 = load_frag(kbase, Dc, kt * 16, 0,  lane);
    v16h kf1 = load_frag(kbase, Dc, kt * 16, 32, lane);
    v8f c = {};
    c = wmma_f32_f16(qf0, kf0, c);
    c = wmma_f32_f16(qf1, kf1, c);
#pragma unroll
    for (int e = 0; e < 8; e++)
      sc[(e + 8 * g) * Sc + kt * 16 + col] = c[e] * 0.125f;  // 1/sqrt(64)
  }
  __syncthreads();

  if (lane < 16) {                                     // exact softmax per row
    float* row = sc + lane * Sc;
    float mx = -3.0e38f;
    for (int j = 0; j < Sc; j++) mx = fmaxf(mx, row[j]);
    float s = 0.0f;
    for (int j = 0; j < Sc; j++) { float e = __expf(row[j] - mx); row[j] = e; s += e; }
    const float inv = 1.0f / s;
    for (int j = 0; j < Sc; j++) row[j] *= inv;
  }
  __syncthreads();

  // emit attention map (the one mandatory HBM stream)
  float* gout = attnL + (size_t)(b * Hc + head) * Sc * Sc + (size_t)(qt * 16) * Sc;
  for (int idx = lane; idx < 16 * Sc; idx += 32) gout[idx] = sc[idx];
  __syncthreads();

  // in-place f32 -> f16 (write window always trails read window)
  _Float16* sh = (_Float16*)sc;
  for (int idx = lane; idx < 16 * Sc; idx += 32) sh[idx] = (_Float16)sc[idx];
  __syncthreads();

  const _Float16* vbase = Vt + (size_t)(b * Hc + head) * HDc * Sc;
#pragma unroll
  for (int nt = 0; nt < 4; nt++) {
    v8f o = {};
    for (int ks = 0; ks < Sc; ks += 32) {
      v16h af  = load_frag(sh,    Sc, 0,       ks, lane);
      v16h bfv = load_frag(vbase, Sc, nt * 16, ks, lane);
      o = wmma_f32_f16(af, bfv, o);
    }
#pragma unroll
    for (int e = 0; e < 8; e++) {
      const int m = qt * 16 + e + 8 * g;
      AO[((size_t)b * Sc + m) * Dc + head * HDc + nt * 16 + col] = (_Float16)o[e];
    }
  }
}

// ---------------------------------------------------------------------------
// small utility kernels
// ---------------------------------------------------------------------------
__global__ __launch_bounds__(256) void convert_transpose(
    const float* __restrict__ W, _Float16* __restrict__ Wt, int K, int N) {
  const size_t idx = (size_t)blockIdx.x * 256 + threadIdx.x;
  if (idx >= (size_t)K * N) return;
  const int n = (int)(idx % N);
  const int k = (int)(idx / N);
  Wt[(size_t)n * K + k] = (_Float16)W[idx];
}

__global__ __launch_bounds__(256) void convert_f16(
    const float* __restrict__ X, _Float16* __restrict__ Y, size_t n) {
  const size_t idx = (size_t)blockIdx.x * 256 + threadIdx.x;
  if (idx < n) Y[idx] = (_Float16)X[idx];
}

__global__ __launch_bounds__(256) void transpose_v(
    const _Float16* __restrict__ V, _Float16* __restrict__ Vt) {
  const size_t idx = (size_t)blockIdx.x * 256 + threadIdx.x;
  if (idx >= (size_t)Bc * Sc * Dc) return;
  const int d = (int)(idx % Dc);
  const size_t bs = idx / Dc;
  const int s = (int)(bs % Sc);
  const int b = (int)(bs / Sc);
  const int head = d / HDc, hd = d % HDc;
  Vt[(((size_t)b * Hc + head) * HDc + hd) * Sc + s] = V[idx];
}

// fused residual + LayerNorm: h = LN(a + h); writes f32 state + f16 mirror
__global__ __launch_bounds__(256) void resid_ln(
    const float* __restrict__ a, float* __restrict__ h32,
    const float* __restrict__ gm, const float* __restrict__ bt,
    _Float16* __restrict__ h16) {
  __shared__ float red[256];
  const int row = blockIdx.x, t = threadIdx.x;
  const size_t base = (size_t)row * Dc;
  const float v0 = a[base + t]       + h32[base + t];
  const float v1 = a[base + t + 256] + h32[base + t + 256];
  red[t] = v0 + v1;
  __syncthreads();
  for (int s = 128; s > 0; s >>= 1) { if (t < s) red[t] += red[t + s]; __syncthreads(); }
  const float mu = red[0] * (1.0f / Dc);
  __syncthreads();
  const float d0 = v0 - mu, d1 = v1 - mu;
  red[t] = d0 * d0 + d1 * d1;
  __syncthreads();
  for (int s = 128; s > 0; s >>= 1) { if (t < s) red[t] += red[t + s]; __syncthreads(); }
  const float rstd = rsqrtf(red[0] * (1.0f / Dc) + 1e-5f);
  const float y0 = d0 * rstd * gm[t]       + bt[t];
  const float y1 = d1 * rstd * gm[t + 256] + bt[t + 256];
  h32[base + t]       = y0;  h32[base + t + 256] = y1;
  h16[base + t]       = (_Float16)y0;
  h16[base + t + 256] = (_Float16)y1;
}

// final heads: q = last@Wfc+bfc ; u = softmax(last@Wu+bu)
__global__ __launch_bounds__(128) void heads_kernel(
    const float* __restrict__ h32, const float* __restrict__ Wfc,
    const float* __restrict__ bfc, const float* __restrict__ Wu,
    const float* __restrict__ bu, float* __restrict__ out) {
  __shared__ float tmp[96];
  const int t = threadIdx.x;
  if (t < 96) {
    const int b = t / 6, r = t % 6, a = r % 3, isU = r / 3;
    const float* last = h32 + ((size_t)b * Sc + (Sc - 1)) * Dc;
    const float* W = isU ? Wu : Wfc;
    float s = isU ? bu[a] : bfc[a];
    for (int d = 0; d < Dc; d++) s += last[d] * W[d * 3 + a];
    tmp[t] = s;
  }
  __syncthreads();
  if (t < 16) {
    out[t * 3 + 0] = tmp[t * 6 + 0];
    out[t * 3 + 1] = tmp[t * 6 + 1];
    out[t * 3 + 2] = tmp[t * 6 + 2];
    const float u0 = tmp[t * 6 + 3], u1 = tmp[t * 6 + 4], u2 = tmp[t * 6 + 5];
    const float m  = fmaxf(u0, fmaxf(u1, u2));
    const float e0 = __expf(u0 - m), e1 = __expf(u1 - m), e2 = __expf(u2 - m);
    const float inv = 1.0f / (e0 + e1 + e2);
    out[48 + t * 3 + 0] = e0 * inv;
    out[48 + t * 3 + 1] = e1 * inv;
    out[48 + t * 3 + 2] = e2 * inv;
  }
}

// ---------------------------------------------------------------------------
extern "C" void kernel_launch(void* const* d_in, const int* in_sizes, int n_in,
                              void* d_out, int out_size, void* d_ws, size_t ws_size,
                              hipStream_t stream) {
  const float* x    = (const float*)d_in[0];
  const float* W_in = (const float*)d_in[1];
  const float* b_in = (const float*)d_in[2];
  const float* Wq   = (const float*)d_in[3];
  const float* bq   = (const float*)d_in[4];
  const float* Wk   = (const float*)d_in[5];
  const float* bk   = (const float*)d_in[6];
  const float* Wv   = (const float*)d_in[7];
  const float* bv   = (const float*)d_in[8];
  const float* Wo   = (const float*)d_in[9];
  const float* bo   = (const float*)d_in[10];
  const float* g1   = (const float*)d_in[11];
  const float* be1  = (const float*)d_in[12];
  const float* g2   = (const float*)d_in[13];
  const float* be2  = (const float*)d_in[14];
  const float* W1   = (const float*)d_in[15];
  const float* b1   = (const float*)d_in[16];
  const float* W2   = (const float*)d_in[17];
  const float* b2   = (const float*)d_in[18];
  const float* Wfc  = (const float*)d_in[19];
  const float* bfc  = (const float*)d_in[20];
  const float* Wu   = (const float*)d_in[21];
  const float* bu   = (const float*)d_in[22];
  float* out = (float*)d_out;

  char* ws = (char*)d_ws;
  size_t off = 0;
  auto take = [&](size_t bytes) -> char* {
    char* p = ws + off;
    off += (bytes + 255) & ~(size_t)255;
    return p;
  };

  float*    h32  = (float*)   take((size_t)Mtok * Dc * 4);
  float*    t32  = (float*)   take((size_t)Mtok * Dc * 4);
  _Float16* h16  = (_Float16*)take((size_t)Mtok * Dc * 2);
  _Float16* q16  = (_Float16*)take((size_t)Mtok * Dc * 2);
  _Float16* k16  = (_Float16*)take((size_t)Mtok * Dc * 2);
  _Float16* v16  = (_Float16*)take((size_t)Mtok * Dc * 2);
  _Float16* vt16 = (_Float16*)take((size_t)Mtok * Dc * 2);
  _Float16* ao16 = (_Float16*)take((size_t)Mtok * Dc * 2);
  _Float16* ff16 = (_Float16*)take((size_t)Mtok * DFFc * 2);
  _Float16* x16  = (_Float16*)take((size_t)Mtok * Fc * 2);
  _Float16* Wtin = (_Float16*)take((size_t)Fc * Dc * 2);
  _Float16* Wqt  = (_Float16*)take((size_t)Lc * Dc * Dc * 2);
  _Float16* Wkt  = (_Float16*)take((size_t)Lc * Dc * Dc * 2);
  _Float16* Wvt  = (_Float16*)take((size_t)Lc * Dc * Dc * 2);
  _Float16* Wot  = (_Float16*)take((size_t)Lc * Dc * Dc * 2);
  _Float16* W1t  = (_Float16*)take((size_t)Lc * Dc * DFFc * 2);
  _Float16* W2t  = (_Float16*)take((size_t)Lc * DFFc * Dc * 2);

  // ---- weight conversion (f32 row-major -> f16 transposed / N-major) ----
  convert_transpose<<<(Fc * Dc + 255) / 256, 256, 0, stream>>>(W_in, Wtin, Fc, Dc);
  for (int l = 0; l < Lc; l++) {
    const size_t wdd = (size_t)l * Dc * Dc;
    const size_t wdf = (size_t)l * Dc * DFFc;
    convert_transpose<<<(Dc * Dc + 255) / 256, 256, 0, stream>>>(Wq + wdd, Wqt + wdd, Dc, Dc);
    convert_transpose<<<(Dc * Dc + 255) / 256, 256, 0, stream>>>(Wk + wdd, Wkt + wdd, Dc, Dc);
    convert_transpose<<<(Dc * Dc + 255) / 256, 256, 0, stream>>>(Wv + wdd, Wvt + wdd, Dc, Dc);
    convert_transpose<<<(Dc * Dc + 255) / 256, 256, 0, stream>>>(Wo + wdd, Wot + wdd, Dc, Dc);
    convert_transpose<<<(Dc * DFFc + 255) / 256, 256, 0, stream>>>(W1 + wdf, W1t + wdf, Dc, DFFc);
    convert_transpose<<<(DFFc * Dc + 255) / 256, 256, 0, stream>>>(W2 + wdf, W2t + wdf, DFFc, Dc);
  }
  convert_f16<<<(int)(((size_t)Mtok * Fc + 255) / 256), 256, 0, stream>>>(
      x, x16, (size_t)Mtok * Fc);

  // ---- input projection: h = relu(x @ W_in + b_in) ----
  gemm_f16_wmma<1, 1, 1><<<dim3(Mtok / 128, Dc / 256), 256, 0, stream>>>(
      x16, Wtin, b_in, h32, h16, Mtok, Dc, Fc);

  const dim3 gDD(Mtok / 128, Dc / 256);      // N=512
  const dim3 gDF(Mtok / 128, DFFc / 256);    // N=1024

  for (int l = 0; l < Lc; l++) {
    const size_t wdd = (size_t)l * Dc * Dc;
    const size_t wdf = (size_t)l * Dc * DFFc;

    gemm_f16_wmma<0, 1, 0><<<gDD, 256, 0, stream>>>(
        h16, Wqt + wdd, bq + (size_t)l * Dc, nullptr, q16, Mtok, Dc, Dc);
    gemm_f16_wmma<0, 1, 0><<<gDD, 256, 0, stream>>>(
        h16, Wkt + wdd, bk + (size_t)l * Dc, nullptr, k16, Mtok, Dc, Dc);
    gemm_f16_wmma<0, 1, 0><<<gDD, 256, 0, stream>>>(
        h16, Wvt + wdd, bv + (size_t)l * Dc, nullptr, v16, Mtok, Dc, Dc);

    transpose_v<<<(int)(((size_t)Mtok * Dc + 255) / 256), 256, 0, stream>>>(v16, vt16);

    attn_wmma<<<dim3(Sc / 16, Hc, Bc), 32, 0, stream>>>(
        q16, k16, vt16,
        out + 96 + (size_t)l * Bc * Hc * Sc * Sc, ao16);

    gemm_f16_wmma<0, 0, 1><<<gDD, 256, 0, stream>>>(
        ao16, Wot + wdd, bo + (size_t)l * Dc, t32, nullptr, Mtok, Dc, Dc);
    resid_ln<<<Mtok, 256, 0, stream>>>(t32, h32, g1 + (size_t)l * Dc,
                                       be1 + (size_t)l * Dc, h16);

    gemm_f16_wmma<1, 1, 0><<<gDF, 256, 0, stream>>>(
        h16, W1t + wdf, b1 + (size_t)l * DFFc, nullptr, ff16, Mtok, DFFc, Dc);
    gemm_f16_wmma<0, 0, 1><<<gDD, 256, 0, stream>>>(
        ff16, W2t + wdf, b2 + (size_t)l * Dc, t32, nullptr, Mtok, Dc, DFFc);
    resid_ln<<<Mtok, 256, 0, stream>>>(t32, h32, g2 + (size_t)l * Dc,
                                       be2 + (size_t)l * Dc, h16);
  }

  heads_kernel<<<1, 128, 0, stream>>>(h32, Wfc, bfc, Wu, bu, out);
}
